// PointNet2_14697378087179
// MI455X (gfx1250) — matrix operand order repaired and checked by reading, not compiled, not run on hardware
//
#include <hip/hip_runtime.h>
#include <hip/hip_bf16.h>
#include <stdint.h>

typedef __attribute__((ext_vector_type(16))) _Float16 v16h;
typedef __attribute__((ext_vector_type(8)))  float    v8f;

#define FPS_T 256

// ---------------- utility kernels ----------------
__global__ void k_zero(float* p, int n) {
  int i = blockIdx.x * blockDim.x + threadIdx.x;
  if (i < n) p[i] = 0.0f;
}
__global__ void k_copy(float* dst, const float* src, int n) {
  int i = blockIdx.x * blockDim.x + threadIdx.x;
  if (i < n) dst[i] = src[i];
}
// pad weight (O,K) row-major -> (Op,Kp) with zeros
__global__ void k_padw(const float* __restrict__ W, float* __restrict__ Wp,
                       int O, int K, int Op, int Kp) {
  int i = blockIdx.x * blockDim.x + threadIdx.x;
  if (i >= Op * Kp) return;
  int o = i / Kp, k = i % Kp;
  Wp[i] = (o < O && k < K) ? W[(size_t)o * K + k] : 0.0f;
}
// (B, Cin, N) channel-major -> (B, N, 3) row-major (first 3 channels)
__global__ void k_xyz_t(const float* __restrict__ in, float* __restrict__ out,
                        int B_, int Cin, int N_) {
  int i = blockIdx.x * blockDim.x + threadIdx.x;
  if (i >= B_ * N_) return;
  int b = i / N_, n = i % N_;
  const float* pb = in + (size_t)b * Cin * N_;
  float* o = out + (size_t)i * 3;
  o[0] = pb[0 * N_ + n];
  o[1] = pb[1 * N_ + n];
  o[2] = pb[2 * N_ + n];
}

// ---------------- farthest point sampling ----------------
__global__ void k_fps(const float* __restrict__ xyz, int N, int S,
                      float* __restrict__ dist, int* __restrict__ idx_out,
                      float* __restrict__ newxyz) {
  int b = blockIdx.x, t = threadIdx.x;
  const float* xb = xyz + (size_t)b * N * 3;
  float* db = dist + (size_t)b * N;
  __shared__ float sv[FPS_T];
  __shared__ int   si[FPS_T];
  int chunk = (N + FPS_T - 1) / FPS_T;
  int n0 = t * chunk;
  int n1 = n0 + chunk; if (n1 > N) n1 = N;
  for (int n = n0; n < n1; ++n) db[n] = 1e10f;
  __syncthreads();
  int far = 0;
  for (int it = 0; it < S; ++it) {
    if (t == 0) {
      idx_out[b * S + it] = far;
      newxyz[((size_t)b * S + it) * 3 + 0] = xb[far * 3 + 0];
      newxyz[((size_t)b * S + it) * 3 + 1] = xb[far * 3 + 1];
      newxyz[((size_t)b * S + it) * 3 + 2] = xb[far * 3 + 2];
    }
    float cx = xb[far * 3 + 0], cy = xb[far * 3 + 1], cz = xb[far * 3 + 2];
    float bm = -1.0f; int bi = 0;
    for (int n = n0; n < n1; ++n) {
      float dx = xb[n * 3 + 0] - cx;
      float dy = xb[n * 3 + 1] - cy;
      float dz = xb[n * 3 + 2] - cz;
      float d = dx * dx + dy * dy + dz * dz;
      float nd = fminf(db[n], d);
      db[n] = nd;
      if (nd > bm) { bm = nd; bi = n; }
    }
    sv[t] = bm; si[t] = bi;
    __syncthreads();
    for (int s2 = FPS_T >> 1; s2 > 0; s2 >>= 1) {
      if (t < s2 && sv[t + s2] > sv[t]) { sv[t] = sv[t + s2]; si[t] = si[t + s2]; }
      __syncthreads();
    }
    far = si[0];
    __syncthreads();
  }
}

// ---------------- ball query: one wave32 per center ----------------
__global__ void k_ballq(const float* __restrict__ xyz, const float* __restrict__ centers,
                        int N, int S, int nW, float r2, int nsample,
                        int* __restrict__ out) {
  int gt = blockIdx.x * blockDim.x + threadIdx.x;
  int gw = gt >> 5, lane = gt & 31;
  if (gw >= nW) return;
  int b = gw / S, s = gw % S;
  const float* xb = xyz + (size_t)b * N * 3;
  const float* c = centers + ((size_t)b * S + s) * 3;
  float cx = c[0], cy = c[1], cz = c[2];
  int* o = out + ((size_t)b * S + s) * nsample;
  int cnt = 0;
  int firstIdx = 0;
  unsigned lmask = (1u << lane) - 1u;
  for (int base = 0; base < N && cnt < nsample; base += 32) {
    int n = base + lane;
    bool ok = false;
    if (n < N) {
      float dx = xb[n * 3 + 0] - cx;
      float dy = xb[n * 3 + 1] - cy;
      float dz = xb[n * 3 + 2] - cz;
      ok = (dx * dx + dy * dy + dz * dz) <= r2;
    }
    unsigned mm = (unsigned)__ballot(ok);
    if (cnt == 0 && mm) firstIdx = base + __builtin_ctz(mm);
    int prefix = __builtin_popcount(mm & lmask);
    if (ok && cnt + prefix < nsample) o[cnt + prefix] = n;
    cnt += __builtin_popcount(mm);
  }
  if (cnt > nsample) cnt = nsample;
  for (int j = lane; j < nsample; j += 32)
    if (j >= cnt) o[j] = firstIdx;
}

// ---------------- SA grouping/gather (writes padded rows, zero tail) -------------
__global__ void k_group(const float* __restrict__ xyz, int Nsrc,
                        const float* __restrict__ pts, int pts_sb, int pts_sn, int pts_sc, int Cp,
                        const float* __restrict__ cen, const int* __restrict__ idx,
                        float* __restrict__ out, int Sn, int Ks, int rows, int ldo) {
  int i = blockIdx.x * blockDim.x + threadIdx.x;
  if (i >= rows) return;
  int bs = i / Ks;
  int b = bs / Sn;
  int n = idx[i];
  const float* xb = xyz + (size_t)b * Nsrc * 3;
  const float* c = cen + (size_t)bs * 3;
  float* o = out + (size_t)i * ldo;
  o[0] = xb[n * 3 + 0] - c[0];
  o[1] = xb[n * 3 + 1] - c[1];
  o[2] = xb[n * 3 + 2] - c[2];
  const float* pb = pts + (size_t)b * pts_sb;
  for (int cc = 0; cc < Cp; ++cc)
    o[3 + cc] = pb[(size_t)cc * pts_sc + (size_t)n * pts_sn];
  for (int cc = 3 + Cp; cc < ldo; ++cc) o[cc] = 0.0f;
}

// ---------------- max over nsample ----------------
__global__ void k_maxpool(const float* __restrict__ in, float* __restrict__ out,
                          int BS, int Ks, int C) {
  int i = blockIdx.x * blockDim.x + threadIdx.x;
  if (i >= BS * C) return;
  int c = i % C, bs = i / C;
  const float* base = in + (size_t)bs * Ks * C + c;
  float m = -1e30f;
  for (int k = 0; k < Ks; ++k) m = fmaxf(m, base[(size_t)k * C]);
  out[i] = m;
}

// ---------------- 3-NN feature interpolation (writes padded rows) ----------------
__global__ void k_interp(const float* __restrict__ xyz1, const float* __restrict__ xyz2,
                         const float* __restrict__ p2, int N2, int C2,
                         const float* __restrict__ p1, int p1_sb, int p1_sn, int p1_sc, int C1,
                         float* __restrict__ out, int N1, int total, int ldo) {
  int i = blockIdx.x * blockDim.x + threadIdx.x;
  if (i >= total) return;
  int b = i / N1, n = i % N1;
  const float* q = xyz1 + (size_t)i * 3;
  const float* x2 = xyz2 + (size_t)b * N2 * 3;
  float qx = q[0], qy = q[1], qz = q[2];
  float bd0 = 1e30f, bd1 = 1e30f, bd2 = 1e30f;
  int i0 = 0, i1 = 0, i2 = 0;
  for (int m = 0; m < N2; ++m) {
    float dx = x2[m * 3 + 0] - qx;
    float dy = x2[m * 3 + 1] - qy;
    float dz = x2[m * 3 + 2] - qz;
    float d = dx * dx + dy * dy + dz * dz;
    if (d < bd0)      { bd2 = bd1; i2 = i1; bd1 = bd0; i1 = i0; bd0 = d; i0 = m; }
    else if (d < bd1) { bd2 = bd1; i2 = i1; bd1 = d;  i1 = m; }
    else if (d < bd2) { bd2 = d;  i2 = m; }
  }
  float r0 = 1.0f / (bd0 + 1e-8f);
  float r1 = 1.0f / (bd1 + 1e-8f);
  float r2 = 1.0f / (bd2 + 1e-8f);
  float rs = r0 + r1 + r2;
  r0 /= rs; r1 /= rs; r2 /= rs;
  float* o = out + (size_t)i * ldo;
  if (p1) {
    const float* pb1 = p1 + (size_t)b * p1_sb;
    for (int c = 0; c < C1; ++c)
      o[c] = pb1[(size_t)c * p1_sc + (size_t)n * p1_sn];
  }
  const float* pb2 = p2 + (size_t)b * N2 * C2;
  const float* a0 = pb2 + (size_t)i0 * C2;
  const float* a1 = pb2 + (size_t)i1 * C2;
  const float* a2 = pb2 + (size_t)i2 * C2;
  for (int c = 0; c < C2; ++c)
    o[C1 + c] = r0 * a0[c] + r1 * a1[c] + r2 * a2[c];
  for (int c = C1 + C2; c < ldo; ++c) o[c] = 0.0f;
}

// ---------------- l4 output transpose (B,S,C) -> (B,C,S) ----------------
__global__ void k_l4_out(const float* __restrict__ l4, float* __restrict__ out,
                         int B_, int S, int C) {
  int i = blockIdx.x * blockDim.x + threadIdx.x;
  if (i >= B_ * S * C) return;
  int b = i / (S * C);
  int rem = i % (S * C);
  int c = rem / S, s = rem % S;
  out[i] = l4[((size_t)b * S + s) * C + c];
}

// ---------------- WMMA GEMM ----------------
// Y[M, ldY] (cols < O stored) = X[M, Kp] * Wp[Op, Kp]^T (+bias).
// Kp % 32 == 0, Op % 32 == 0, M % 64 == 0. X/Wp rows are zero-padded, so the
// K-loop is completely branch-free: 12x global_load_b128 + cvt + 2x v_wmma.
// Block = 128 threads = 4 waves; each wave computes a 16x32 tile (2 acc).
__global__ void k_wmma_gemm(const float* __restrict__ X, const float* __restrict__ Wp,
                            float* __restrict__ Y, int M, int Kp, int O, int ldY,
                            float* stats, const float* bias) {
  int wave = threadIdx.x >> 5;
  int lane = threadIdx.x & 31;
  int hl  = lane >> 4;          // 0: lanes 0-15, 1: lanes 16-31
  int l15 = lane & 15;
  int tm = blockIdx.x * 64 + wave * 16;
  int tn = blockIdx.y * 32;

  // A 16x32 f16, contiguous k order chosen here (not AMD's table order): lane half 0 holds K {0..15}, half 1 holds {16..31}
  const float* xrow  = X  + (size_t)(tm + l15) * Kp + hl * 16;
  // B 32x16 f16, same contiguous k order as A: half j <-> K, lanes 0-15 K 0-15, lanes 16-31 K 16-31
  const float* wrow0 = Wp + (size_t)(tn + l15) * Kp + hl * 16;
  const float* wrow1 = Wp + (size_t)(tn + 16 + l15) * Kp + hl * 16;

  v8f acc0 = {}, acc1 = {};
  for (int kb = 0; kb < Kp; kb += 32) {
    __builtin_prefetch(xrow + kb + 32, 0, 1);   // global_prefetch_b8
    __builtin_prefetch(wrow0 + kb + 32, 0, 1);
    float4 xa0 = *(const float4*)(xrow + kb);
    float4 xa1 = *(const float4*)(xrow + kb + 4);
    float4 xa2 = *(const float4*)(xrow + kb + 8);
    float4 xa3 = *(const float4*)(xrow + kb + 12);
    float4 wb0 = *(const float4*)(wrow0 + kb);
    float4 wb1 = *(const float4*)(wrow0 + kb + 4);
    float4 wb2 = *(const float4*)(wrow0 + kb + 8);
    float4 wb3 = *(const float4*)(wrow0 + kb + 12);
    float4 wc0 = *(const float4*)(wrow1 + kb);
    float4 wc1 = *(const float4*)(wrow1 + kb + 4);
    float4 wc2 = *(const float4*)(wrow1 + kb + 8);
    float4 wc3 = *(const float4*)(wrow1 + kb + 12);
    v16h a, b0, b1;
    a[0]=(_Float16)xa0.x;  a[1]=(_Float16)xa0.y;  a[2]=(_Float16)xa0.z;  a[3]=(_Float16)xa0.w;
    a[4]=(_Float16)xa1.x;  a[5]=(_Float16)xa1.y;  a[6]=(_Float16)xa1.z;  a[7]=(_Float16)xa1.w;
    a[8]=(_Float16)xa2.x;  a[9]=(_Float16)xa2.y;  a[10]=(_Float16)xa2.z; a[11]=(_Float16)xa2.w;
    a[12]=(_Float16)xa3.x; a[13]=(_Float16)xa3.y; a[14]=(_Float16)xa3.z; a[15]=(_Float16)xa3.w;
    b0[0]=(_Float16)wb0.x;  b0[1]=(_Float16)wb0.y;  b0[2]=(_Float16)wb0.z;  b0[3]=(_Float16)wb0.w;
    b0[4]=(_Float16)wb1.x;  b0[5]=(_Float16)wb1.y;  b0[6]=(_Float16)wb1.z;  b0[7]=(_Float16)wb1.w;
    b0[8]=(_Float16)wb2.x;  b0[9]=(_Float16)wb2.y;  b0[10]=(_Float16)wb2.z; b0[11]=(_Float16)wb2.w;
    b0[12]=(_Float16)wb3.x; b0[13]=(_Float16)wb3.y; b0[14]=(_Float16)wb3.z; b0[15]=(_Float16)wb3.w;
    b1[0]=(_Float16)wc0.x;  b1[1]=(_Float16)wc0.y;  b1[2]=(_Float16)wc0.z;  b1[3]=(_Float16)wc0.w;
    b1[4]=(_Float16)wc1.x;  b1[5]=(_Float16)wc1.y;  b1[6]=(_Float16)wc1.z;  b1[7]=(_Float16)wc1.w;
    b1[8]=(_Float16)wc2.x;  b1[9]=(_Float16)wc2.y;  b1[10]=(_Float16)wc2.z; b1[11]=(_Float16)wc2.w;
    b1[12]=(_Float16)wc3.x; b1[13]=(_Float16)wc3.y; b1[14]=(_Float16)wc3.z; b1[15]=(_Float16)wc3.w;
    acc0 = __builtin_amdgcn_wmma_f32_16x16x32_f16(false, a, false, b0,
                                                  (short)0, acc0, false, false);
    acc1 = __builtin_amdgcn_wmma_f32_16x16x32_f16(false, a, false, b1,
                                                  (short)0, acc1, false, false);
  }

  // C/D layout: VGPR r -> row = tm + r + 8*hl; col = tn(+16) + (lane&15)
  int col0 = tn + l15;
  int col1 = tn + 16 + l15;
  float s0 = 0.0f, q0 = 0.0f, s1 = 0.0f, q1 = 0.0f;
#pragma unroll
  for (int r = 0; r < 8; ++r) {
    int row = tm + r + hl * 8;
    float v0 = acc0[r];
    float v1 = acc1[r];
    if (bias) {
      if (col0 < O) v0 += bias[col0];
      if (col1 < O) v1 += bias[col1];
    }
    if (col0 < O) { Y[(size_t)row * ldY + col0] = v0; s0 += v0; q0 += v0 * v0; }
    if (col1 < O) { Y[(size_t)row * ldY + col1] = v1; s1 += v1; q1 += v1 * v1; }
  }
  if (stats) {
    s0 += __shfl_down(s0, 16, 32);
    q0 += __shfl_down(q0, 16, 32);
    s1 += __shfl_down(s1, 16, 32);
    q1 += __shfl_down(q1, 16, 32);
    if (hl == 0) {
      if (col0 < O) { atomicAdd(&stats[col0], s0); atomicAdd(&stats[256 + col0], q0); }
      if (col1 < O) { atomicAdd(&stats[col1], s1); atomicAdd(&stats[256 + col1], q1); }
    }
  }
}

// ---------------- batch-norm (population) + ReLU / leaky ----------------
__global__ void k_bn_act(float* __restrict__ Y, int M, int O,
                         const float* __restrict__ stats, float slope) {
  int i = blockIdx.x * blockDim.x + threadIdx.x;
  long total = (long)M * O;
  if (i >= total) return;
  int c = i % O;
  float inv = 1.0f / (float)M;
  float mean = stats[c] * inv;
  float var = stats[256 + c] * inv - mean * mean;
  float v = (Y[i] - mean) * rsqrtf(var + 1e-5f);
  Y[i] = v >= 0.0f ? v : v * slope;
}

// ====================================================================
extern "C" void kernel_launch(void* const* d_in, const int* in_sizes, int n_in,
                              void* d_out, int out_size, void* d_ws, size_t ws_size,
                              hipStream_t stream) {
  (void)in_sizes; (void)n_in; (void)out_size; (void)ws_size;
  const int B = 8, N = 16384, CIN = 35;
  const int S1 = 128, K1 = 32, S2 = 64, K2 = 32;
  const int R1 = B * S1 * K1;     // 32768 rows (SA1 MLP)
  const int R2 = B * S2 * K2;     // 16384 rows (SA2 MLP)
  const int RF2 = B * S1;         // 1024 rows (FP2 MLP)
  const int RF1 = B * N;          // 131072 rows (FP1/conv MLP)

  const float* xyz = (const float*)d_in[0];
  const float* conv2b = (const float*)d_in[14];
  float* out = (float*)d_out;

  // ---- workspace carve-up ----
  char* ws = (char*)d_ws;
  size_t cur = 0;
  auto alloc = [&](size_t bytes) {
    void* p = ws + cur;
    cur = (cur + bytes + 255) & ~(size_t)255;
    return p;
  };
  float* xyz_t     = (float*)alloc((size_t)B * N * 3 * 4);
  float* fps_dist  = (float*)alloc((size_t)B * N * 4);
  int*   fps_idx1  = (int*)  alloc((size_t)B * S1 * 4);
  float* new_xyz1  = (float*)alloc((size_t)B * S1 * 3 * 4);
  int*   ball1     = (int*)  alloc((size_t)B * S1 * K1 * 4);
  int*   fps_idx2  = (int*)  alloc((size_t)B * S2 * 4);
  float* new_xyz2  = (float*)alloc((size_t)B * S2 * 3 * 4);
  int*   ball2     = (int*)  alloc((size_t)B * S2 * K2 * 4);
  float* l1_pts    = (float*)alloc((size_t)B * S1 * 64 * 4);
  float* l4_buf    = (float*)alloc((size_t)B * S2 * 128 * 4);
  float* l1_new    = (float*)alloc((size_t)B * S1 * 128 * 4);
  float* statsA    = (float*)alloc((size_t)16 * 512 * 4);
  float* wpadA     = (float*)alloc((size_t)13 * 65536 * 4);  // padded weights
  float* bufA      = (float*)alloc((size_t)RF1 * 128 * 4);   // 64 MB
  float* bufB      = (float*)alloc((size_t)RF1 * 128 * 4);   // 64 MB

  auto wp = [&](int slot) { return wpadA + (size_t)slot * 65536; };
  auto st = [&](int slot) { return statsA + slot * 512; };

  auto padw = [&](int slot, int din_idx, int O, int K, int Op, int Kp) {
    int tot = Op * Kp;
    k_padw<<<dim3((tot + 255) / 256), dim3(256), 0, stream>>>(
        (const float*)d_in[din_idx], wp(slot), O, K, Op, Kp);
  };
  auto gemm = [&](const float* X, const float* Wp_, float* Y, int M, int Kp,
                  int O, int Op, int ldY, float* stv, const float* bias) {
    dim3 g((unsigned)(M / 64), (unsigned)(Op / 32));
    k_wmma_gemm<<<g, dim3(128), 0, stream>>>(X, Wp_, Y, M, Kp, O, ldY, stv, bias);
  };
  auto bnact = [&](float* Y, int M, int O, const float* stv, float slope) {
    long total = (long)M * O;
    k_bn_act<<<dim3((unsigned)((total + 255) / 256)), dim3(256), 0, stream>>>(Y, M, O, stv, slope);
  };

  // 0) zero stats arena, pad all weights to (ceil32(O), ceil32(K))
  k_zero<<<dim3(32), dim3(256), 0, stream>>>(statsA, 16 * 512);
  padw(0,  1, 32, 38, 32, 64);     // sa1_w0
  padw(1,  2, 32, 32, 32, 32);     // sa1_w1
  padw(2,  3, 64, 32, 64, 32);     // sa1_w2
  padw(3,  4, 64, 67, 64, 96);     // sa2_w0
  padw(4,  5, 64, 64, 64, 64);     // sa2_w1
  padw(5,  6, 128, 64, 128, 64);   // sa2_w2
  padw(6,  7, 256, 192, 256, 192); // fp2_w0
  padw(7,  8, 128, 256, 128, 256); // fp2_w1
  padw(8,  9, 128, 128, 128, 128); // fp1_w0
  padw(9, 10, 128, 128, 128, 128); // fp1_w1
  padw(10,11, 128, 128, 128, 128); // fp1_w2
  padw(11,12, 128, 128, 128, 128); // conv1_w
  padw(12,13, 1, 128, 32, 128);    // conv2_w

  // 1) xyz -> (B,N,3)
  k_xyz_t<<<dim3((B * N + 255) / 256), dim3(256), 0, stream>>>(xyz, xyz_t, B, CIN, N);

  // 2) SA1: FPS + ball query + gather (rows padded to Kp=64)
  k_fps<<<dim3(B), dim3(FPS_T), 0, stream>>>(xyz_t, N, S1, fps_dist, fps_idx1, new_xyz1);
  k_ballq<<<dim3(B * S1 * 32 / 256), dim3(256), 0, stream>>>(
      xyz_t, new_xyz1, N, S1, B * S1, 0.09f, K1, ball1);
  k_group<<<dim3((R1 + 255) / 256), dim3(256), 0, stream>>>(
      xyz_t, N, xyz, CIN * N, 1, N, CIN, new_xyz1, ball1, bufA, S1, K1, R1, 64);

  // 3) SA1 MLP: 38->32->32->64
  gemm(bufA, wp(0), bufB, R1, 64, 32, 32, 32, st(0), nullptr); bnact(bufB, R1, 32, st(0), 0.0f);
  gemm(bufB, wp(1), bufA, R1, 32, 32, 32, 32, st(1), nullptr); bnact(bufA, R1, 32, st(1), 0.0f);
  gemm(bufA, wp(2), bufB, R1, 32, 64, 64, 64, st(2), nullptr); bnact(bufB, R1, 64, st(2), 0.0f);
  k_maxpool<<<dim3((B * S1 * 64 + 255) / 256), dim3(256), 0, stream>>>(
      bufB, l1_pts, B * S1, K1, 64);

  // 4) SA2: FPS over (B,128,3) + ball query + gather (rows padded to Kp=96)
  k_fps<<<dim3(B), dim3(FPS_T), 0, stream>>>(new_xyz1, S1, S2, fps_dist, fps_idx2, new_xyz2);
  k_ballq<<<dim3(B * S2 * 32 / 256), dim3(256), 0, stream>>>(
      new_xyz1, new_xyz2, S1, S2, B * S2, 0.36f, K2, ball2);
  k_group<<<dim3((R2 + 255) / 256), dim3(256), 0, stream>>>(
      new_xyz1, S1, l1_pts, S1 * 64, 64, 1, 64, new_xyz2, ball2, bufA, S2, K2, R2, 96);

  // 5) SA2 MLP: 67->64->64->128
  gemm(bufA, wp(3), bufB, R2, 96, 64, 64, 64, st(3), nullptr);    bnact(bufB, R2, 64, st(3), 0.0f);
  gemm(bufB, wp(4), bufA, R2, 64, 64, 64, 64, st(4), nullptr);    bnact(bufA, R2, 64, st(4), 0.0f);
  gemm(bufA, wp(5), bufB, R2, 64, 128, 128, 128, st(5), nullptr); bnact(bufB, R2, 128, st(5), 0.0f);
  k_maxpool<<<dim3((B * S2 * 128 + 255) / 256), dim3(256), 0, stream>>>(
      bufB, l4_buf, B * S2, K2, 128);

  // 6) second output: l4_points (B,128,64) channel-major
  k_l4_out<<<dim3((B * S2 * 128 + 255) / 256), dim3(256), 0, stream>>>(
      l4_buf, out + (size_t)B * N, B, S2, 128);

  // 7) FP2: 3-NN interp of l4 onto l1 positions; [l1_pts(64) | interp(128)] = 192 ch
  k_interp<<<dim3((RF2 + 255) / 256), dim3(256), 0, stream>>>(
      new_xyz1, new_xyz2, l4_buf, S2, 128,
      l1_pts, S1 * 64, 64, 1, 64, bufA, S1, RF2, 192);
  gemm(bufA, wp(6), bufB, RF2, 192, 256, 256, 256, st(6), nullptr); bnact(bufB, RF2, 256, st(6), 0.0f);
  gemm(bufB, wp(7), bufA, RF2, 256, 128, 128, 128, st(7), nullptr); bnact(bufA, RF2, 128, st(7), 0.0f);
  k_copy<<<dim3((B * S1 * 128 + 255) / 256), dim3(256), 0, stream>>>(
      l1_new, bufA, B * S1 * 128);

  // 8) FP1: 3-NN interp of l1_new onto all 16384 points
  k_interp<<<dim3((RF1 + 255) / 256), dim3(256), 0, stream>>>(
      xyz_t, new_xyz1, l1_new, S1, 128,
      nullptr, 0, 0, 0, 0, bufA, N, RF1, 128);
  gemm(bufA, wp(8),  bufB, RF1, 128, 128, 128, 128, st(8),  nullptr); bnact(bufB, RF1, 128, st(8),  0.0f);
  gemm(bufB, wp(9),  bufA, RF1, 128, 128, 128, 128, st(9),  nullptr); bnact(bufA, RF1, 128, st(9),  0.0f);
  gemm(bufA, wp(10), bufB, RF1, 128, 128, 128, 128, st(10), nullptr); bnact(bufB, RF1, 128, st(10), 0.0f);

  // 9) conv1 (BN + leaky 0.01), conv2 (bias only, O=1) -> first output region (B,1,N)
  gemm(bufB, wp(11), bufA, RF1, 128, 128, 128, 128, st(11), nullptr); bnact(bufA, RF1, 128, st(11), 0.01f);
  gemm(bufA, wp(12), out, RF1, 128, 1, 32, 1, nullptr, conv2b);
}
